// Attention_82566451299127
// MI455X (gfx1250) — compile-verified, matrix operand order checked by reading, not run on hardware
//
#include <hip/hip_runtime.h>
#include <hip/hip_bf16.h>

// Problem constants (match reference)
#define B_ 4
#define T_ 1024
#define C_ 768
#define H_ 12
#define DH_ 64
#define BH_ (B_ * H_)          // 48
#define BT_ (B_ * T_)          // 4096

typedef _Float16     v16h __attribute__((ext_vector_type(16)));
typedef float        v8f  __attribute__((ext_vector_type(8)));
typedef unsigned int v4u  __attribute__((ext_vector_type(4)));
typedef int          v8i  __attribute__((ext_vector_type(8)));
typedef int          v4i  __attribute__((ext_vector_type(4)));

#if __has_builtin(__builtin_amdgcn_tensor_load_to_lds)
#define HAVE_TDM 1
#else
#define HAVE_TDM 0
#endif

// ---------------------------------------------------------------------------
// WMMA helper (CDNA5: V_WMMA_F32_16X16X32_F16, wave32)
// ---------------------------------------------------------------------------
__device__ __forceinline__ v8f wmma_f16(v16h a, v16h b, v8f c) {
  // 8 args: (neg_a, A, neg_b, B, c_mod, C, reuse_a, reuse_b)
  return __builtin_amdgcn_wmma_f32_16x16x32_f16(false, a, false, b, (short)0, c,
                                                false, false);
}

// Load one lane's 16-half A/B fragment for a 16x32 (or 32x16) f16 tile.
// Per ISA layout: element j<8  -> K = hi*8 + j ; j>=8 -> K = 16 + hi*8 + (j-8)
// i.e. two contiguous 8-half (16B) runs at p+hi*8 and p+16+hi*8.
__device__ __forceinline__ v16h ldfrag(const _Float16* __restrict__ p, int hi) {
  v16h r;
  const _Float16* p0 = p + hi * 8;
#pragma unroll
  for (int j = 0; j < 8; ++j) r[j] = p0[j];
#pragma unroll
  for (int j = 0; j < 8; ++j) r[8 + j] = p0[16 + j];
  return r;
}

// ---------------------------------------------------------------------------
// Tensor Data Mover: 2D tile (2-byte elements) global -> LDS.
// D# built per CDNA5 ISA ch.8: group0 {count=1, lds_addr, global_addr, type=2},
// group1 {data_size=2B, tensor_dim0/1, tile_dim0/1, dim0_stride}. Groups 2/3
// zero (2D tile; tile_dim2=0 means unused).
// ---------------------------------------------------------------------------
__device__ __forceinline__ unsigned lds_off_of(const void* p) {
  // Generic pointers to LDS carry the LDS byte offset in addr[31:0].
  return (unsigned)(unsigned long long)p;
}

__device__ __forceinline__ void tdm_load_2d(unsigned lds_byte_off,
                                            const void* gptr,
                                            unsigned width_elems,
                                            unsigned rows,
                                            unsigned stride_elems) {
#if HAVE_TDM
  unsigned long long ga = (unsigned long long)gptr;
  v4u g0;
  g0[0] = 1u;                                            // count=1 (user D#)
  g0[1] = lds_byte_off;                                  // lds_addr (bytes)
  g0[2] = (unsigned)(ga & 0xFFFFFFFFu);                  // global_addr[31:0]
  g0[3] = (unsigned)((ga >> 32) & 0x01FFFFFFu) | (2u << 30);  // [56:32] | type=2
  v8i g1;
  g1[0] = (int)(1u << 16);                               // data_size = 2 bytes
  g1[1] = (int)((width_elems & 0xFFFFu) << 16);          // tensor_dim0[15:0]
  g1[2] = (int)(((width_elems >> 16) & 0xFFFFu) |
                ((rows & 0xFFFFu) << 16));               // dim0 hi | dim1 lo
  g1[3] = (int)(((rows >> 16) & 0xFFFFu) |
                ((width_elems & 0xFFFFu) << 16));        // dim1 hi | tile_dim0
  g1[4] = (int)(rows & 0xFFFFu);                         // tile_dim1 (tile_dim2=0)
  g1[5] = (int)stride_elems;                             // dim0_stride[31:0]
  g1[6] = 0;                                             // stride0 hi | stride1 lo
  g1[7] = 0;
  v4i gz = {0, 0, 0, 0};
#if __clang_major__ >= 23
  v8i gz8 = {0, 0, 0, 0, 0, 0, 0, 0};
  __builtin_amdgcn_tensor_load_to_lds(g0, g1, gz, gz, gz8, 0);
#else
  __builtin_amdgcn_tensor_load_to_lds(g0, g1, gz, gz, 0);
#endif
#else
  (void)lds_byte_off; (void)gptr; (void)width_elems; (void)rows; (void)stride_elems;
#endif
}

__device__ __forceinline__ void tdm_wait() {
#if __has_builtin(__builtin_amdgcn_s_wait_tensorcnt)
  __builtin_amdgcn_s_wait_tensorcnt(0);
#endif
}

// ---------------------------------------------------------------------------
// Conversion kernels
// ---------------------------------------------------------------------------
__global__ void cvt_f32_to_f16(const float* __restrict__ x,
                               _Float16* __restrict__ y, int n) {
  int i = blockIdx.x * 256 + threadIdx.x;
  if (i < n) y[i] = (_Float16)x[i];
}

// WT[n*768 + k] = (f16) W[k*768 + n]  (transpose: B-fragments contiguous in K)
__global__ void cvt_transpose_f16(const float* __restrict__ W,
                                  _Float16* __restrict__ WT) {
  int i = blockIdx.x * 256 + threadIdx.x;   // over 768*768
  int n = i / C_, k = i % C_;
  WT[i] = (_Float16)W[k * C_ + n];
}

__global__ void init_count(float* __restrict__ count) {
  int i = blockIdx.x * 256 + threadIdx.x;   // over T*T
  count[i] = 0.095f;                        // 0.1 * COUNT_DECAY
}

// ---------------------------------------------------------------------------
// Fused QKV projection: Y = X @ W + b. A band (16x768 f16, 24KB) staged into
// LDS once per block via TDM; one wave per 16x16 tile over a 128-col band.
// blockIdx.z: 0=Q (scaled 1/sqrt(DH)), 1=K, 2=V (stored per-head [d][t]).
// ---------------------------------------------------------------------------
__global__ __launch_bounds__(256) void qkv_gemm_kernel(
    const _Float16* __restrict__ h16,   // [BT, C]
    const _Float16* __restrict__ WT3,   // 3 x [n-major][k]
    const float* __restrict__ bq, const float* __restrict__ bk,
    const float* __restrict__ bv,
    _Float16* __restrict__ q16,         // [B,H,T,DH]
    _Float16* __restrict__ k16,         // [B,H,T,DH]
    _Float16* __restrict__ vT16) {      // [B,H,DH,T]
  __shared__ _Float16 aT[16 * C_];      // 24 KB staged A band

  const int mat  = blockIdx.z;
  const int wave = threadIdx.x >> 5;
  const int lane = threadIdx.x & 31;
  const int hi   = lane >> 4, ln = lane & 15;
  const int row0 = blockIdx.y * 16;
  const int col0 = blockIdx.x * 128 + wave * 16;
  const int bcol = col0 + ln;
  const _Float16* Wm = WT3 + (size_t)mat * (C_ * C_);

#if HAVE_TDM
  if (wave == 0) {   // one wave issues the DMA (EXEC ignored by TDM)
    tdm_load_2d(lds_off_of(aT), h16 + (size_t)row0 * C_, C_, 16, C_);
    tdm_wait();
  }
#else
  for (int i = threadIdx.x; i < 16 * C_; i += 256)
    aT[i] = h16[(size_t)row0 * C_ + i];
#endif
  __syncthreads();

  v8f c = {};
#pragma unroll 4
  for (int k = 0; k < C_; k += 32) {
    v16h a = ldfrag(aT + ln * C_ + k, hi);               // from LDS
    v16h b = ldfrag(Wm + (size_t)bcol * C_ + k, hi);     // from global
    __builtin_prefetch(Wm + (size_t)bcol * C_ + k + 256, 0, 0);
    c = wmma_f16(a, b, c);
  }

  const float* bias = (mat == 0) ? bq : (mat == 1) ? bk : bv;
  const float bval  = bias[bcol];
  const float scale = (mat == 0) ? 0.125f : 1.0f;        // 1/sqrt(64) into Q
  const int hh = bcol >> 6, d = bcol & 63;

#pragma unroll
  for (int r = 0; r < 8; ++r) {
    const int m = r + hi * 8;              // C layout: VGPR r -> M=r (+8 if hi)
    const int grow = row0 + m;
    const int bb = grow >> 10, t = grow & 1023;
    const float val = (c[r] + bval) * scale;
    if (mat == 2) {
      vT16[(((size_t)bb * H_ + hh) * DH_ + d) * T_ + t] = (_Float16)val;
    } else {
      _Float16* dst = (mat == 0) ? q16 : k16;
      dst[(((size_t)bb * H_ + hh) * T_ + t) * DH_ + d] = (_Float16)val;
    }
  }
}

// ---------------------------------------------------------------------------
// Attention: per (b,h, 16-query tile). 256 threads = 8 waves.
//   stage 1: scores (16x1024) via WMMA into 64KB LDS (Q fragments hoisted)
//   stage 2: per-row softmax + bisection top-p (each wave owns 2 full rows)
//   stage 3: ctx = attn @ V via WMMA (4 waves, one 16-wide d-tile each)
// ---------------------------------------------------------------------------
__global__ __launch_bounds__(256) void attn_kernel(
    const _Float16* __restrict__ q16, const _Float16* __restrict__ k16,
    const _Float16* __restrict__ vT16, _Float16* __restrict__ ctx16,
    float* __restrict__ count, const int* __restrict__ counter_p) {
  __shared__ float sm[16 * T_];           // 64 KB score/attn tile

  const int bh = blockIdx.x;              // 0..47
  const int b = bh / H_, h = bh % H_;
  const int q0 = blockIdx.y * 16;
  const int wave = threadIdx.x >> 5;
  const int lane = threadIdx.x & 31;
  const int hi = lane >> 4, ln = lane & 15;

  const _Float16* Q = q16 + (size_t)bh * T_ * DH_;
  const _Float16* K = k16 + (size_t)bh * T_ * DH_;
  const _Float16* V = vT16 + (size_t)bh * DH_ * T_;

  // ---- stage 1: scores = (Q/8) @ K^T, wave w covers keys [w*128, w*128+128)
  const _Float16* qrow = Q + (size_t)(q0 + ln) * DH_;
  const v16h aq0 = ldfrag(qrow, hi);      // loop-invariant Q fragments
  const v16h aq1 = ldfrag(qrow + 32, hi);
#pragma unroll
  for (int tkt = 0; tkt < 8; ++tkt) {
    const int key0 = wave * 128 + tkt * 16;
    const _Float16* krow = K + (size_t)(key0 + ln) * DH_;
    v8f c = {};
    c = wmma_f16(aq0, ldfrag(krow, hi), c);
    c = wmma_f16(aq1, ldfrag(krow + 32, hi), c);
#pragma unroll
    for (int r = 0; r < 8; ++r)
      sm[(r + hi * 8) * T_ + key0 + ln] = c[r];
  }
  __syncthreads();

  // ---- stage 2: softmax + top-p (UCB confidence is constant -> cancels)
  const float topp = 0.9f + 0.1f * __expf(-(float)(*counter_p) / 5000.0f);

  for (int rr = 0; rr < 2; ++rr) {
    const int m = wave * 2 + rr;
    float* row = sm + m * T_;

    float mx = -1e30f;
    for (int i = lane; i < T_; i += 32) mx = fmaxf(mx, row[i]);
#pragma unroll
    for (int off = 16; off; off >>= 1) mx = fmaxf(mx, __shfl_xor(mx, off, 32));

    float s = 0.0f;
    for (int i = lane; i < T_; i += 32) {
      float e = __expf(row[i] - mx);
      row[i] = e;
      s += e;
    }
#pragma unroll
    for (int off = 16; off; off >>= 1) s += __shfl_xor(s, off, 32);

    // bisection: F(tau)=sum{e>tau}; key selected iff F(e_k) <= topp*s
    const float target = topp * s;
    float blo = 0.0f, bhi = 1.0f;         // e in (0, 1]
#pragma unroll 1
    for (int it = 0; it < 24; ++it) {
      const float mid = 0.5f * (blo + bhi);
      float f = 0.0f;
      for (int i = lane; i < T_; i += 32) {
        const float e = row[i];
        f += (e > mid) ? e : 0.0f;
      }
#pragma unroll
      for (int off = 16; off; off >>= 1) f += __shfl_xor(f, off, 32);
      if (f > target) blo = mid; else bhi = mid;
    }

    float ssum = 0.0f;
    for (int i = lane; i < T_; i += 32) {
      const float e = row[i];
      ssum += (e > blo) ? e : 0.0f;
    }
#pragma unroll
    for (int off = 16; off; off >>= 1) ssum += __shfl_xor(ssum, off, 32);

    const float inv = 1.0f / (ssum + 1e-8f * s);  // ref's +1e-8 on probs
    const int gq = q0 + m;
    for (int i = lane; i < T_; i += 32) {
      const float e = row[i];
      const bool sel = e > blo;
      row[i] = sel ? e * inv : 0.0f;
      if (sel)
        atomicAdd(&count[(size_t)gq * T_ + i], 1.0f / (float)BH_);
    }
  }
  __syncthreads();

  // ---- stage 3: ctx(16 x 64) = attn(16 x 1024) @ V(1024 x 64)
  if (wave < 4) {
    const int nt = wave;                  // 16-wide slice of DH
    v8f c = {};
    for (int k = 0; k < T_; k += 32) {
      v16h a;
      const float* ar = sm + ln * T_ + k; // A row = lane&15, f32->f16 on the fly
#pragma unroll
      for (int j = 0; j < 8; ++j) a[j] = (_Float16)ar[hi * 8 + j];
#pragma unroll
      for (int j = 0; j < 8; ++j) a[8 + j] = (_Float16)ar[16 + hi * 8 + j];
      // B element (K=key, n=d): vT16 is [d][t] -> contiguous in key
      v16h bb = ldfrag(V + (size_t)(nt * 16 + ln) * T_ + k, hi);
      c = wmma_f16(a, bb, c);
    }
#pragma unroll
    for (int r = 0; r < 8; ++r) {
      const int m = r + hi * 8;
      const int t = q0 + m;
      ctx16[((size_t)b * T_ + t) * C_ + h * DH_ + nt * 16 + ln] = (_Float16)c[r];
    }
  }
}

// ---------------------------------------------------------------------------
// Output projection: out = ctx @ Wo + bo (f32 output); A band staged via TDM.
// ---------------------------------------------------------------------------
__global__ __launch_bounds__(256) void out_gemm_kernel(
    const _Float16* __restrict__ ctx16, const _Float16* __restrict__ WoT,
    const float* __restrict__ bo, float* __restrict__ out) {
  __shared__ _Float16 aT[16 * C_];        // 24 KB staged ctx band

  const int wave = threadIdx.x >> 5;
  const int lane = threadIdx.x & 31;
  const int hi = lane >> 4, ln = lane & 15;
  const int row0 = blockIdx.y * 16;
  const int col0 = blockIdx.x * 128 + wave * 16;
  const int bcol = col0 + ln;

#if HAVE_TDM
  if (wave == 0) {
    tdm_load_2d(lds_off_of(aT), ctx16 + (size_t)row0 * C_, C_, 16, C_);
    tdm_wait();
  }
#else
  for (int i = threadIdx.x; i < 16 * C_; i += 256)
    aT[i] = ctx16[(size_t)row0 * C_ + i];
#endif
  __syncthreads();

  v8f c = {};
#pragma unroll 4
  for (int k = 0; k < C_; k += 32) {
    v16h a = ldfrag(aT + ln * C_ + k, hi);
    v16h b = ldfrag(WoT + (size_t)bcol * C_ + k, hi);
    __builtin_prefetch(WoT + (size_t)bcol * C_ + k + 256, 0, 0);
    c = wmma_f16(a, b, c);
  }
  const float bval = bo[bcol];
#pragma unroll
  for (int r = 0; r < 8; ++r) {
    const int m = r + hi * 8;
    out[(size_t)(row0 + m) * C_ + bcol] = c[r] + bval;
  }
}

// ---------------------------------------------------------------------------
// Host-side launch
// ---------------------------------------------------------------------------
extern "C" void kernel_launch(void* const* d_in, const int* in_sizes, int n_in,
                              void* d_out, int out_size, void* d_ws,
                              size_t ws_size, hipStream_t stream) {
  (void)in_sizes; (void)n_in; (void)out_size; (void)ws_size;

  const float* hidden = (const float*)d_in[0];
  const float* Wq = (const float*)d_in[1];
  const float* bq = (const float*)d_in[2];
  const float* Wk = (const float*)d_in[3];
  const float* bk = (const float*)d_in[4];
  const float* Wv = (const float*)d_in[5];
  const float* bv = (const float*)d_in[6];
  const float* Wo = (const float*)d_in[7];
  const float* bo = (const float*)d_in[8];
  const int* counter = (const int*)d_in[9];
  // d_in[10] (ucb) unused: with counter=2000 the UCB branch is always taken.

  float* out = (float*)d_out;
  float* count = out + (size_t)BT_ * C_;   // final_count_score [T,T]

  // Workspace carve-up (~34.5 MB total)
  char* p = (char*)d_ws;
  _Float16* h16  = (_Float16*)p; p += (size_t)BT_ * C_ * 2;
  _Float16* WT3  = (_Float16*)p; p += (size_t)3 * C_ * C_ * 2;
  _Float16* WoT  = (_Float16*)p; p += (size_t)C_ * C_ * 2;
  _Float16* q16  = (_Float16*)p; p += (size_t)BT_ * C_ * 2;
  _Float16* k16  = (_Float16*)p; p += (size_t)BT_ * C_ * 2;
  _Float16* vT16 = (_Float16*)p; p += (size_t)BT_ * C_ * 2;
  _Float16* ctx16 = (_Float16*)p;

  const int nh = BT_ * C_;                 // 3,145,728
  const int nw = C_ * C_;                  // 589,824

  cvt_f32_to_f16<<<(nh + 255) / 256, 256, 0, stream>>>(hidden, h16, nh);
  cvt_transpose_f16<<<nw / 256, 256, 0, stream>>>(Wq, WT3 + 0 * (size_t)nw);
  cvt_transpose_f16<<<nw / 256, 256, 0, stream>>>(Wk, WT3 + 1 * (size_t)nw);
  cvt_transpose_f16<<<nw / 256, 256, 0, stream>>>(Wv, WT3 + 2 * (size_t)nw);
  cvt_transpose_f16<<<nw / 256, 256, 0, stream>>>(Wo, WoT);

  qkv_gemm_kernel<<<dim3(C_ / 128, BT_ / 16, 3), 256, 0, stream>>>(
      h16, WT3, bq, bk, bv, q16, k16, vT16);

  init_count<<<(T_ * T_) / 256, 256, 0, stream>>>(count);

  attn_kernel<<<dim3(BH_, T_ / 16), 256, 0, stream>>>(q16, k16, vT16, ctx16,
                                                      count, counter);

  out_gemm_kernel<<<dim3(C_ / 128, BT_ / 16), 256, 0, stream>>>(ctx16, WoT, bo,
                                                                out);
}
